// Attention_60129542720
// MI455X (gfx1250) — matrix operand order repaired and checked by reading, not compiled, not run on hardware
//
#include <hip/hip_runtime.h>
#include <math.h>

typedef unsigned short u16;
typedef unsigned int   u32;
typedef __attribute__((ext_vector_type(16))) __bf16 bf16x16;
typedef __attribute__((ext_vector_type(8)))  float  f32x8;
typedef __attribute__((ext_vector_type(4)))  int    i32x4;

#define P_TOT 262144   // B*H*W
#define HW    65536    // H*W

#if defined(__has_builtin)
#  if __has_builtin(__builtin_amdgcn_global_load_async_to_lds_b128)
#    define HAVE_ASYNC_LDS 1
#  endif
#  if __has_builtin(__builtin_amdgcn_s_wait_asynccnt)
#    define HAVE_WAIT_ASYNC 1
#  endif
#endif

typedef __attribute__((address_space(1))) i32x4 g_i32x4;
typedef __attribute__((address_space(3))) i32x4 l_i32x4;

// 16B global -> LDS copy; async (ASYNCcnt, no VGPR staging) when available.
__device__ inline void lds_copy_b128(void* l, const void* g) {
#ifdef HAVE_ASYNC_LDS
    __builtin_amdgcn_global_load_async_to_lds_b128((g_i32x4*)g, (l_i32x4*)l, 0, 0);
#else
    *(uint4*)l = *(const uint4*)g;
#endif
}

// make async LDS writes visible block-wide
__device__ inline void async_fence_block() {
#ifdef HAVE_ASYNC_LDS
#  ifdef HAVE_WAIT_ASYNC
    __builtin_amdgcn_s_wait_asynccnt(0);
#  else
    asm volatile("s_wait_asynccnt 0x0" ::: "memory");
#  endif
#endif
    __syncthreads();
}

union FragAB {
    bf16x16 v;
    uint4 q[2];
};

__device__ inline u16 f2bf(float f) {
    u32 u = __float_as_uint(f);
    u32 r = u + 0x7fffu + ((u >> 16) & 1u);
    return (u16)(r >> 16);
}

__device__ inline f32x8 wmma_bf16(const FragAB& a, const FragAB& b, f32x8 c) {
    return __builtin_amdgcn_wmma_f32_16x16x32_bf16(false, a.v, false, b.v, (short)0, c, false, false);
}

// ---------------- small prep kernels ----------------

__global__ void k_zero_stats(float* stats) {
    if (threadIdx.x < 32) stats[threadIdx.x] = 0.f;
}

// X (B,C,H,W) f32 -> Xb (P,C) bf16, P index = b*HW + hw
__global__ void k_pack_x(const float* __restrict__ X, u16* __restrict__ Xb) {
    int t = blockIdx.x * 256 + threadIdx.x;       // t < 96*P
    int c = t >> 18;
    int p = t & (P_TOT - 1);
    int b = p >> 16, hw = p & 65535;
    Xb[p * 96 + c] = f2bf(X[(b * 96 + c) * HW + hw]);
}

__global__ void k_pack_wqkv(const float* __restrict__ QK_w, const float* __restrict__ QK_b,
                            const float* __restrict__ V_w,  const float* __restrict__ V_b,
                            u16* __restrict__ Wq, float* __restrict__ biasAll) {
    int t = blockIdx.x * 256 + threadIdx.x;
    if (t < 288 * 96) {
        int d = t / 96, c = t % 96;
        float v = d < 192 ? QK_w[d * 96 + c] : V_w[(d - 192) * 96 + c];
        Wq[t] = f2bf(v);
    }
    if (t < 288) biasAll[t] = (t < 192) ? QK_b[t] : V_b[t - 192];
}

__global__ void k_pack_bf16(const float* __restrict__ src, u16* __restrict__ dst, int n) {
    int t = blockIdx.x * 256 + threadIdx.x;
    if (t < n) dst[t] = f2bf(src[t]);
}

// relative position bias table: bt[h][n][m], h<6, n,m<64
__global__ void k_bias_tab(const float* __restrict__ mw1, const float* __restrict__ mb1,
                           const float* __restrict__ mw2, const float* __restrict__ mb2,
                           float* __restrict__ bt) {
    int t = blockIdx.x * 256 + threadIdx.x;
    if (t >= 4096) return;
    int n = t >> 6, m = t & 63;
    float dy = (float)((n >> 3) - (m >> 3));
    float dx = (float)((n & 7) - (m & 7));
    float sy = ((dy > 0.f) - (dy < 0.f)) * log1pf(fabsf(dy));
    float sx = ((dx > 0.f) - (dx < 0.f)) * log1pf(fabsf(dx));
    float o[6];
    for (int h = 0; h < 6; h++) o[h] = mb2[h];
    for (int r = 0; r < 256; r++) {
        float hv = mw1[2 * r] * sy + mw1[2 * r + 1] * sx + mb1[r];
        hv = hv > 0.f ? hv : 0.f;
        for (int h = 0; h < 6; h++) o[h] += mw2[h * 256 + r] * hv;
    }
    for (int h = 0; h < 6; h++) bt[h * 4096 + t] = o[h];
}

// ---------------- GEMM1: QKV projection ----------------
// M=P (16/block), N=288, K=96.  Output scattered to window layout (q pre-scaled),
// V channels also stored NCHW f32 for the depthwise conv.
__global__ void __launch_bounds__(256) k_gemm_qkv(const u16* __restrict__ Xb, const u16* __restrict__ Wq,
                                                  const float* __restrict__ biasAll,
                                                  u16* __restrict__ QKVwin, float* __restrict__ Vn) {
    __shared__ __align__(16) u16 Bs[288 * 96];
    __shared__ __align__(16) u16 As[16 * 96];
    __shared__ float bb[288];
    int tid = threadIdx.x;
    for (int i = tid; i < 3456; i += 256) lds_copy_b128(Bs + i * 8, Wq + i * 8);
    int pbase = blockIdx.x * 16;
    if (tid < 192) lds_copy_b128(As + tid * 8, Xb + (size_t)pbase * 96 + tid * 8);
    for (int i = tid; i < 288; i += 256) bb[i] = biasAll[i];
    async_fence_block();

    int wave = tid >> 5, lane = tid & 31, g = lane >> 4, ln = lane & 15;
    // per-block pixel-row constants (16 consecutive pixels share b_ and h)
    int b_ = pbase >> 16;
    int hw0 = pbase & 65535;
    int h = hw0 >> 8, w0 = hw0 & 255;
    int winbase = b_ * 1024 + (h >> 3) * 32;
    int nrow = (h & 7) << 3;

    for (int t = wave; t < 18; t += 8) {
        int n0 = t * 16;
        f32x8 acc = {0.f, 0.f, 0.f, 0.f, 0.f, 0.f, 0.f, 0.f};
        for (int kk = 0; kk < 96; kk += 32) {
            FragAB a, b;
            int ab = ln * 96 + kk + g * 8;
            a.q[0] = *(const uint4*)&As[ab];
            a.q[1] = *(const uint4*)&As[ab + 16];
            int bo = (n0 + ln) * 96 + kk + g * 8;
            b.q[0] = *(const uint4*)&Bs[bo];
            b.q[1] = *(const uint4*)&Bs[bo + 16];
            acc = wmma_bf16(a, b, acc);
        }
        int n0s = __builtin_amdgcn_readfirstlane(n0);   // tile id is wave-uniform
        float sc = (n0s < 96) ? 0.25f : 1.0f;           // q pre-scale (hd^-0.5)
        int d = n0 + ln;
        float bias = bb[d];
        for (int r = 0; r < 8; r++) {
            int w = w0 + r + g * 8;
            float val = acc[r] + bias;
            int win = winbase + (w >> 3);
            int n = nrow + (w & 7);
            QKVwin[(win * 64 + n) * 288 + d] = f2bf(val * sc);
            if (n0s >= 192)
                Vn[(b_ * 96 + (d - 192)) * HW + ((h << 8) + w)] = val;
        }
    }
}

// ---------------- window attention (one wave per window-head) ----------------
__global__ void __launch_bounds__(32) k_win_attn(const u16* __restrict__ QKVwin, const float* __restrict__ bt,
                                                 float* __restrict__ aw, u16* __restrict__ awb) {
    __shared__ __align__(16) float attf[64 * 64];
    __shared__ __align__(16) u16 attb[64 * 64];
    __shared__ __align__(16) u16 vT[16 * 64];
    int win = blockIdx.x, h = blockIdx.y;
    int lane = threadIdx.x, g = lane >> 4, ln = lane & 15;
    const u16* base = QKVwin + win * 64 * 288;

    for (int i = lane; i < 1024; i += 32) {
        int tok = i & 63, hd = i >> 6;
        vT[hd * 64 + tok] = base[tok * 288 + 192 + h * 16 + hd];
    }

    f32x8 zacc = {0.f, 0.f, 0.f, 0.f, 0.f, 0.f, 0.f, 0.f};
    f32x8 acc[4][4];
    for (int i = 0; i < 4; i++) for (int j = 0; j < 4; j++) acc[i][j] = zacc;
    uint4 uz = make_uint4(0u, 0u, 0u, 0u);

    // attn = (q*scale) @ k^T     (K=16 zero-padded to 32)
    for (int tm = 0; tm < 4; tm++) {
        FragAB a;
        int tokn = tm * 16 + ln;
        a.q[0] = *(const uint4*)(base + tokn * 288 + h * 16 + g * 8);
        a.q[1] = uz;
        for (int tn = 0; tn < 4; tn++) {
            FragAB b;
            b.q[0] = *(const uint4*)(base + (tn * 16 + ln) * 288 + 96 + h * 16 + g * 8);
            b.q[1] = uz;
            acc[tm][tn] = wmma_bf16(a, b, acc[tm][tn]);
        }
    }
    const float* bth = bt + h * 4096;
    for (int tm = 0; tm < 4; tm++)
        for (int tn = 0; tn < 4; tn++)
            for (int r = 0; r < 8; r++) {
                int row = tm * 16 + r + g * 8;
                int col = tn * 16 + ln;
                attf[row * 64 + col] = acc[tm][tn][r] + bth[row * 64 + col];
            }
    __syncthreads();

    for (int row = lane; row < 64; row += 32) {
        float mx = -1e30f;
        for (int j = 0; j < 64; j++) mx = fmaxf(mx, attf[row * 64 + j]);
        float s = 0.f;
        for (int j = 0; j < 64; j++) { float e = __expf(attf[row * 64 + j] - mx); attf[row * 64 + j] = e; s += e; }
        float inv = 1.f / s;
        for (int j = 0; j < 64; j++) attb[row * 64 + j] = f2bf(attf[row * 64 + j] * inv);
    }
    __syncthreads();

    // out = attn @ v   (M=64, N=16, K=64)
    for (int tm = 0; tm < 4; tm++) {
        f32x8 acc2 = zacc;
        for (int kk = 0; kk < 64; kk += 32) {
            FragAB a, b;
            int ab = (tm * 16 + ln) * 64 + kk + g * 8;
            a.q[0] = *(const uint4*)&attb[ab];
            a.q[1] = *(const uint4*)&attb[ab + 16];
            int bo = ln * 64 + kk + g * 8;
            b.q[0] = *(const uint4*)&vT[bo];
            b.q[1] = *(const uint4*)&vT[bo + 16];
            acc2 = wmma_bf16(a, b, acc2);
        }
        for (int r = 0; r < 8; r++) {
            int tok = tm * 16 + r + g * 8;
            int idx = (win * 64 + tok) * 96 + h * 16 + ln;
            float v = acc2[r];
            aw[idx] = v;
            awb[idx] = f2bf(v);
        }
    }
}

// ---------------- statistics ----------------
__global__ void k_src_stats(const float* __restrict__ aw, float* stats) {
    __shared__ float s1[256], s2[256];
    int blk = blockIdx.x, batch = blk >> 10;
    int base = batch * 6291456 + (blk & 1023) * 6144;
    float a = 0.f, b = 0.f;
    for (int j = threadIdx.x; j < 6144; j += 256) { float v = aw[base + j]; a += v; b += v * v; }
    s1[threadIdx.x] = a; s2[threadIdx.x] = b;
    __syncthreads();
    for (int st = 128; st > 0; st >>= 1) {
        if (threadIdx.x < st) { s1[threadIdx.x] += s1[threadIdx.x + st]; s2[threadIdx.x] += s2[threadIdx.x + st]; }
        __syncthreads();
    }
    if (threadIdx.x == 0) { atomicAdd(&stats[batch], s1[0]); atomicAdd(&stats[4 + batch], s2[0]); }
}

// ---------------- pe MLP hidden: relu(xg @ pe_w1^T + b1), M=4096 K=6144 N=256 ----------------
__global__ void __launch_bounds__(256) k_pe_hidden(const u16* __restrict__ awb, const u16* __restrict__ pw1b,
                                                   const float* __restrict__ pb1, float* __restrict__ hidden) {
    __shared__ __align__(16) u16 As[16 * 32];
    __shared__ __align__(16) u16 Bs[256 * 32];
    int tid = threadIdx.x;
    int wave = tid >> 5, lane = tid & 31, g = lane >> 4, ln = lane & 15;
    int mbase = blockIdx.x * 16;
    f32x8 zacc = {0.f, 0.f, 0.f, 0.f, 0.f, 0.f, 0.f, 0.f};
    f32x8 acc[2]; acc[0] = zacc; acc[1] = zacc;
    for (int kk = 0; kk < 6144; kk += 32) {
        __syncthreads();
        if (tid < 64) {
            int row = tid >> 2, seg = tid & 3;
            lds_copy_b128(As + row * 32 + seg * 8,
                          awb + (size_t)(mbase + row) * 6144 + kk + seg * 8);
        }
        for (int i = tid; i < 1024; i += 256) {
            int row = i >> 2, seg = i & 3;
            const u16* gp = pw1b + (size_t)row * 6144 + kk + seg * 8;
            lds_copy_b128(Bs + row * 32 + seg * 8, gp);
            if (kk + 32 < 6144) __builtin_prefetch(gp + 32, 0, 1);
        }
        async_fence_block();
        FragAB a;
        a.q[0] = *(const uint4*)&As[ln * 32 + g * 8];
        a.q[1] = *(const uint4*)&As[ln * 32 + g * 8 + 16];
        for (int tt = 0; tt < 2; tt++) {
            int n = wave * 32 + tt * 16 + ln;
            FragAB b;
            b.q[0] = *(const uint4*)&Bs[n * 32 + g * 8];
            b.q[1] = *(const uint4*)&Bs[n * 32 + g * 8 + 16];
            acc[tt] = wmma_bf16(a, b, acc[tt]);
        }
    }
    for (int tt = 0; tt < 2; tt++) {
        int n = wave * 32 + tt * 16 + ln;
        float bias = pb1[n];
        for (int r = 0; r < 8; r++) {
            int m = mbase + r + g * 8;
            float v = acc[tt][r] + bias;
            hidden[m * 256 + n] = v > 0.f ? v : 0.f;
        }
    }
}

// emb = l2norm(hidden @ pe_w2^T + b2); accumulate cur-stats
__global__ void k_emb(const float* __restrict__ hidden, const float* __restrict__ pw2,
                      const float* __restrict__ pb2, float* __restrict__ emb, float* stats) {
    __shared__ float sh[32];
    int win = blockIdx.x, t = threadIdx.x;
    float e = 0.f;
    if (t < 30) {
        e = pb2[t];
        const float* hrow = hidden + win * 256;
        const float* wrow = pw2 + t * 256;
        for (int r = 0; r < 256; r++) e += hrow[r] * wrow[r];
    }
    sh[t] = (t < 30) ? e * e : 0.f;
    __syncthreads();
    if (t == 0) { float s = 0.f; for (int i = 0; i < 30; i++) s += sh[i]; sh[31] = sqrtf(s) + 1e-8f; }
    __syncthreads();
    if (t < 30) {
        float v = e / sh[31];
        emb[win * 30 + t] = v;
        int batch = win >> 10;
        atomicAdd(&stats[8 + batch], v);
        atomicAdd(&stats[12 + batch], v * v);
    }
}

__global__ void k_finalize(float* stats) {
    int t = threadIdx.x;
    if (t < 4) {
        float c1 = 1.f / 6291456.f;
        float m = stats[t] * c1;
        float v = stats[4 + t] * c1 - m * m;
        stats[16 + t] = m; stats[20 + t] = sqrtf(v + 1e-8f);
        float c2 = 1.f / 30720.f;
        float m2 = stats[8 + t] * c2;
        float v2 = stats[12 + t] * c2 - m2 * m2;
        stats[24 + t] = m2; stats[28 + t] = sqrtf(v2 + 1e-8f);
    }
}

// att_g raw = exp(30*alpha*<emb_n,emb_m> + 20*beta)
__global__ void k_attg(const float* __restrict__ emb, const float* __restrict__ alpha,
                       const float* __restrict__ beta, float* __restrict__ attg) {
    int t = blockIdx.x * 256 + threadIdx.x;
    int b = t >> 20, rem = t & 1048575;
    int n = rem >> 10, m = rem & 1023;
    const float* en = emb + (b * 1024 + n) * 30;
    const float* em = emb + (b * 1024 + m) * 30;
    float d = 0.f;
    for (int i = 0; i < 30; i++) d += en[i] * em[i];
    attg[t] = __expf(30.f * alpha[0] * d + beta[0] * 20.f);
}

__global__ void k_attg_norm(const float* __restrict__ attg, u16* __restrict__ attgn) {
    __shared__ float s1[256];
    int row = blockIdx.x;
    const float* src = attg + row * 1024;
    float a = 0.f;
    for (int j = threadIdx.x; j < 1024; j += 256) a += src[j];
    s1[threadIdx.x] = a;
    __syncthreads();
    for (int st = 128; st > 0; st >>= 1) {
        if (threadIdx.x < st) s1[threadIdx.x] += s1[threadIdx.x + st];
        __syncthreads();
    }
    float inv = 1.f / (s1[0] + 1e-8f);
    for (int j = threadIdx.x; j < 1024; j += 256) attgn[row * 1024 + j] = f2bf(src[j] * inv);
}

// rw = att_gn @ xg   per batch: M=1024, K=1024, N=6144
__global__ void __launch_bounds__(256) k_rw(const u16* __restrict__ attgn, const u16* __restrict__ awb,
                                            float* __restrict__ rw) {
    __shared__ __align__(16) u16 As[64 * 32];
    __shared__ __align__(16) u16 Bs[64 * 32];
    int tid = threadIdx.x, wave = tid >> 5, lane = tid & 31, g = lane >> 4, ln = lane & 15;
    int nbase = blockIdx.x * 64, mbase = blockIdx.y * 64, b = blockIdx.z;
    int mi = wave & 3, ni = wave >> 2;
    f32x8 zacc = {0.f, 0.f, 0.f, 0.f, 0.f, 0.f, 0.f, 0.f};
    f32x8 acc[2]; acc[0] = zacc; acc[1] = zacc;
    const u32* aw32 = (const u32*)awb;
    for (int kk = 0; kk < 1024; kk += 32) {
        __syncthreads();
        {   // A tile: verbatim rows -> async copy
            int row = tid >> 2, seg = tid & 3;
            lds_copy_b128(As + row * 32 + seg * 8,
                          attgn + (size_t)(b * 1024 + mbase + row) * 1024 + kk + seg * 8);
        }
        for (int i = tid; i < 1024; i += 256) {   // B tile: 16-bit transpose via DS scatter
            int kr = i >> 5, cc = i & 31;
            size_t gi = ((size_t)(b * 1024 + kk + kr) * 6144 + nbase) / 2 + cc;
            u32 v = aw32[gi];
            Bs[(2 * cc) * 32 + kr] = (u16)(v & 0xffffu);
            Bs[(2 * cc + 1) * 32 + kr] = (u16)(v >> 16);
            if (kk + 32 < 1024) __builtin_prefetch(aw32 + gi + 32 * 3072, 0, 1);
        }
        async_fence_block();
        FragAB a;
        int ar = mi * 16 + ln;
        a.q[0] = *(const uint4*)&As[ar * 32 + g * 8];
        a.q[1] = *(const uint4*)&As[ar * 32 + g * 8 + 16];
        for (int tt = 0; tt < 2; tt++) {
            int n = ni * 32 + tt * 16 + ln;
            FragAB bf;
            bf.q[0] = *(const uint4*)&Bs[n * 32 + g * 8];
            bf.q[1] = *(const uint4*)&Bs[n * 32 + g * 8 + 16];
            acc[tt] = wmma_bf16(a, bf, acc[tt]);
        }
    }
    for (int tt = 0; tt < 2; tt++)
        for (int r = 0; r < 8; r++) {
            int m = mbase + mi * 16 + r + g * 8;
            int n = nbase + ni * 32 + tt * 16 + ln;
            rw[(b * 1024 + m) * 6144 + n] = acc[tt][r];
        }
}

// normalize rw + window-reverse + depthwise 5x5 (reflect) + pack proj input bf16
__global__ void k_fuse(const float* __restrict__ rw, const float* __restrict__ Vn,
                       const float* __restrict__ dww, const float* __restrict__ dwb,
                       const float* __restrict__ stats, u16* __restrict__ ProjIn) {
    int t = blockIdx.x * 256 + threadIdx.x;
    int c = t >> 18;
    int p = t & (P_TOT - 1);
    int b = p >> 16, hw = p & 65535, h = hw >> 8, w = hw & 255;
    const float* vch = Vn + (b * 96 + c) * HW;
    const float* wk = dww + c * 25;
    float conv = dwb[c];
    for (int ky = 0; ky < 5; ky++) {
        int rh = h + ky - 2; rh = rh < 0 ? -rh : (rh > 255 ? 510 - rh : rh);
        for (int kx = 0; kx < 5; kx++) {
            int rwm = w + kx - 2; rwm = rwm < 0 ? -rwm : (rwm > 255 ? 510 - rwm : rwm);
            conv += vch[rh * 256 + rwm] * wk[ky * 5 + kx];
        }
    }
    int winb = (h >> 3) * 32 + (w >> 3);
    int n = ((h & 7) << 3) + (w & 7);
    float av = rw[(b * 1024 + winb) * 6144 + n * 96 + c];
    av = (av - stats[24 + b]) / stats[28 + b];
    av = av * stats[20 + b] + stats[16 + b];
    ProjIn[p * 96 + c] = f2bf(conv + av);
}

// final projection GEMM: M=P, K=96, N=96, output scattered to NCHW f32
__global__ void __launch_bounds__(256) k_proj(const u16* __restrict__ ProjIn, const u16* __restrict__ pwb,
                                              const float* __restrict__ pb, float* __restrict__ out) {
    __shared__ __align__(16) u16 Bs[96 * 96];
    __shared__ __align__(16) u16 As[16 * 96];
    __shared__ float bb[96];
    int tid = threadIdx.x;
    for (int i = tid; i < 1152; i += 256) lds_copy_b128(Bs + i * 8, pwb + i * 8);
    int pbase = blockIdx.x * 16;
    if (tid < 192) lds_copy_b128(As + tid * 8, ProjIn + (size_t)pbase * 96 + tid * 8);
    if (tid < 96) bb[tid] = pb[tid];
    async_fence_block();

    int wave = tid >> 5, lane = tid & 31, g = lane >> 4, ln = lane & 15;
    if (wave < 6) {
        int n0 = wave * 16;
        f32x8 acc = {0.f, 0.f, 0.f, 0.f, 0.f, 0.f, 0.f, 0.f};
        for (int kk = 0; kk < 96; kk += 32) {
            FragAB a, b;
            int ab = ln * 96 + kk + g * 8;
            a.q[0] = *(const uint4*)&As[ab];
            a.q[1] = *(const uint4*)&As[ab + 16];
            int bo = (n0 + ln) * 96 + kk + g * 8;
            b.q[0] = *(const uint4*)&Bs[bo];
            b.q[1] = *(const uint4*)&Bs[bo + 16];
            acc = wmma_bf16(a, b, acc);
        }
        int d = n0 + ln;
        float bias = bb[d];
        for (int r = 0; r < 8; r++) {
            int p = pbase + r + g * 8;
            int b_ = p >> 16, hw = p & 65535;
            out[(b_ * 96 + d) * HW + hw] = acc[r] + bias;
        }
    }
}

extern "C" void kernel_launch(void* const* d_in, const int* in_sizes, int n_in,
                              void* d_out, int out_size, void* d_ws, size_t ws_size,
                              hipStream_t stream) {
    (void)in_sizes; (void)n_in; (void)out_size; (void)ws_size;
    const float* X        = (const float*)d_in[0];
    const float* V_w      = (const float*)d_in[1];
    const float* V_b      = (const float*)d_in[2];
    const float* QK_w     = (const float*)d_in[3];
    const float* QK_b     = (const float*)d_in[4];
    const float* proj_w   = (const float*)d_in[5];
    const float* proj_b   = (const float*)d_in[6];
    const float* dw_w     = (const float*)d_in[7];
    const float* dw_b     = (const float*)d_in[8];
    const float* meta_w1  = (const float*)d_in[9];
    const float* meta_b1  = (const float*)d_in[10];
    const float* meta_w2  = (const float*)d_in[11];
    const float* meta_b2  = (const float*)d_in[12];
    const float* pe_w1    = (const float*)d_in[13];
    const float* pe_b1    = (const float*)d_in[14];
    const float* pe_w2    = (const float*)d_in[15];
    const float* pe_b2    = (const float*)d_in[16];
    const float* att_alpha= (const float*)d_in[17];
    const float* att_beta = (const float*)d_in[18];

    char* ws = (char*)d_ws;
    size_t off = 0;
    auto alloc = [&](size_t bytes) -> void* {
        void* r = ws + off;
        off += (bytes + 255) & ~(size_t)255;
        return r;
    };
    u16*   Xb      = (u16*)  alloc((size_t)P_TOT * 96 * 2);
    u16*   Wq      = (u16*)  alloc((size_t)288 * 96 * 2);
    float* biasAll = (float*)alloc(288 * 4);
    u16*   pw1b    = (u16*)  alloc((size_t)256 * 6144 * 2);
    u16*   projwb  = (u16*)  alloc((size_t)96 * 96 * 2);
    u16*   QKVwin  = (u16*)  alloc((size_t)4096 * 64 * 288 * 2);
    float* Vn      = (float*)alloc((size_t)4 * 96 * HW * 4);
    float* bt      = (float*)alloc((size_t)6 * 4096 * 4);
    float* aw      = (float*)alloc((size_t)4096 * 64 * 96 * 4);
    u16*   awb     = (u16*)  alloc((size_t)4096 * 64 * 96 * 2);
    float* hidden  = (float*)alloc((size_t)4096 * 256 * 4);
    float* emb     = (float*)alloc((size_t)4096 * 30 * 4);
    float* attg    = (float*)alloc((size_t)4 * 1024 * 1024 * 4);
    u16*   attgn   = (u16*)  alloc((size_t)4 * 1024 * 1024 * 2);
    float* rwbuf   = (float*)alloc((size_t)4096 * 6144 * 4);
    u16*   ProjIn  = (u16*)  alloc((size_t)P_TOT * 96 * 2);
    float* stats   = (float*)alloc(128);

    k_zero_stats<<<1, 32, 0, stream>>>(stats);
    k_pack_x<<<98304, 256, 0, stream>>>(X, Xb);
    k_pack_wqkv<<<108, 256, 0, stream>>>(QK_w, QK_b, V_w, V_b, Wq, biasAll);
    k_pack_bf16<<<6144, 256, 0, stream>>>(pe_w1, pw1b, 256 * 6144);
    k_pack_bf16<<<36, 256, 0, stream>>>(proj_w, projwb, 96 * 96);
    k_bias_tab<<<16, 256, 0, stream>>>(meta_w1, meta_b1, meta_w2, meta_b2, bt);
    k_gemm_qkv<<<16384, 256, 0, stream>>>(Xb, Wq, biasAll, QKVwin, Vn);
    k_win_attn<<<dim3(4096, 6), 32, 0, stream>>>(QKVwin, bt, aw, awb);
    k_src_stats<<<4096, 256, 0, stream>>>(aw, stats);
    k_pe_hidden<<<256, 256, 0, stream>>>(awb, pw1b, pe_b1, hidden);
    k_emb<<<4096, 32, 0, stream>>>(hidden, pe_w2, pe_b2, emb, stats);
    k_finalize<<<1, 4, 0, stream>>>(stats);
    k_attg<<<16384, 256, 0, stream>>>(emb, att_alpha, att_beta, attg);
    k_attg_norm<<<4096, 256, 0, stream>>>(attg, attgn);
    k_rw<<<dim3(96, 16, 4), 256, 0, stream>>>(attgn, awb, rwbuf);
    k_fuse<<<98304, 256, 0, stream>>>(rwbuf, Vn, dw_w, dw_b, stats, ProjIn);
    k_proj<<<16384, 256, 0, stream>>>(ProjIn, projwb, proj_b, (float*)d_out);
}
